// GLM47AttentionKernel_59622736003262
// MI455X (gfx1250) — matrix operand order repaired and checked by reading, not compiled, not run on hardware
//
#include <hip/hip_runtime.h>

typedef __attribute__((ext_vector_type(16))) __bf16 v16bf;
typedef __attribute__((ext_vector_type(8)))  __bf16 v8bf;
typedef __attribute__((ext_vector_type(4)))  __bf16 v4bf;
typedef __attribute__((ext_vector_type(8)))  float  v8f;

namespace {
constexpr int kB = 2, kH = 16, kS = 2048, kD = 128;
constexpr int KT     = 32;            // keys per tile (fills WMMA K=32 for PV)
constexpr int NTILES = kS / KT;       // 64
constexpr int WAVES  = 8;             // 256 threads / block
constexpr int QW     = 16;            // query rows per wave
constexpr int QBLK   = WAVES * QW;    // 128 query rows per block
constexpr int KSTR   = kD + 8;        // 136 bf16 -> 272 B rows (16B aligned)
constexpr int VSTR   = KT + 8;        // 40 bf16  -> 80 B rows  (16B aligned)
constexpr int PSTR   = KT;            // 32 bf16  -> 64 B rows
constexpr float SCALE = 0.08838834764831845f;   // 1/sqrt(128)
constexpr float LOG2E = 1.4426950408889634f;
}

// xor-butterfly within 16-lane rows via v_permlane16_b32 (VALU, no DS wait)
__device__ __forceinline__ float permlane16_xor(float x, unsigned s0, unsigned s1) {
  int xi = __builtin_bit_cast(int, x);
  int r  = __builtin_amdgcn_permlane16(xi, xi, s0, s1, false, false);
  return __builtin_bit_cast(float, r);
}
__device__ __forceinline__ float rowmax16(float x) {
  x = fmaxf(x, permlane16_xor(x, 0x67452301u, 0xEFCDAB89u));  // xor 1
  x = fmaxf(x, permlane16_xor(x, 0x54761032u, 0xDCFE98BAu));  // xor 2
  x = fmaxf(x, permlane16_xor(x, 0x32107654u, 0xBA98FEDCu));  // xor 4
  x = fmaxf(x, permlane16_xor(x, 0xFEDCBA98u, 0x76543210u));  // xor 8
  return x;
}
__device__ __forceinline__ float rowsum16(float x) {
  x += permlane16_xor(x, 0x67452301u, 0xEFCDAB89u);
  x += permlane16_xor(x, 0x54761032u, 0xDCFE98BAu);
  x += permlane16_xor(x, 0x32107654u, 0xBA98FEDCu);
  x += permlane16_xor(x, 0xFEDCBA98u, 0x76543210u);
  return x;
}

__global__ __launch_bounds__(256)
void fattn_bf16_wmma(const float* __restrict__ Q, const float* __restrict__ K,
                     const float* __restrict__ V, const float* __restrict__ Msk,
                     float* __restrict__ O) {
  __shared__ __attribute__((aligned(16))) __bf16 lk[KT * KSTR];          // [key][d]
  __shared__ __attribute__((aligned(16))) __bf16 lvT[kD * VSTR];         // [d][key]
  __shared__ __attribute__((aligned(16))) __bf16 lp[WAVES][QW * PSTR];   // per-wave P

  const int tid  = threadIdx.x;
  const int lane = tid & 31;
  const int w    = tid >> 5;
  const int n    = lane & 15;
  const int half = lane >> 4;

  const int bh    = blockIdx.x / (kS / QBLK);
  const int qblk  = (blockIdx.x % (kS / QBLK)) * QBLK;
  const size_t hb = (size_t)bh * kS * kD;
  const int qwave = qblk + w * QW;

  // ---- Q A-fragments (16x32 bf16 x 4 chunks), softmax scale folded in ----
  v16bf qf[4];
  {
    const float* qrow = Q + hb + (size_t)(qwave + n) * kD;
    #pragma unroll
    for (int c = 0; c < 4; ++c) {
      const float* p0 = qrow + 32 * c + half * 16;  // contiguous k, same as kf: run at half*16
      const float* p1 = p0 + 8;                     // and half*16+8
      #pragma unroll
      for (int j = 0; j < 8; ++j) {
        qf[c][j]     = (__bf16)(p0[j] * SCALE);
        qf[c][j + 8] = (__bf16)(p1[j] * SCALE);
      }
    }
  }

  v8f acc[8];
  #pragma unroll
  for (int dc = 0; dc < 8; ++dc) acc[dc] = {};
  float mrow[8], lrow[8];
  #pragma unroll
  for (int r = 0; r < 8; ++r) { mrow[r] = -__builtin_inff(); lrow[r] = 0.f; }

  for (int kt = 0; kt < NTILES; ++kt) {
    const int k0 = kt * KT;

    // ---- cooperative stage: K tile -> lk (bf16), V tile -> lvT (bf16, transposed)
    #pragma unroll
    for (int i = 0; i < 4; ++i) {
      const int idx = tid + i * 256;        // 1024 float4 per tile
      const int key = idx >> 5;             // 32 float4 per key row
      const int d4  = (idx & 31) * 4;
      const float4 kv = *(const float4*)(K + hb + (size_t)(k0 + key) * kD + d4);
      v4bf kb = { (__bf16)kv.x, (__bf16)kv.y, (__bf16)kv.z, (__bf16)kv.w };
      *(v4bf*)&lk[key * KSTR + d4] = kb;
      const float4 vv = *(const float4*)(V + hb + (size_t)(k0 + key) * kD + d4);
      lvT[(d4 + 0) * VSTR + key] = (__bf16)vv.x;
      lvT[(d4 + 1) * VSTR + key] = (__bf16)vv.y;
      lvT[(d4 + 2) * VSTR + key] = (__bf16)vv.z;
      lvT[(d4 + 3) * VSTR + key] = (__bf16)vv.w;
    }
    if (kt + 1 < NTILES) {  // pull next tile toward L2 (global_prefetch_b8)
      const int key = tid >> 5;
      const int d4  = (tid & 31) * 4;
      __builtin_prefetch(K + hb + (size_t)(k0 + KT + key) * kD + d4, 0, 1);
      __builtin_prefetch(V + hb + (size_t)(k0 + KT + key) * kD + d4, 0, 1);
    }
    __syncthreads();

    // ---- preload all K B-fragments, then back-to-back QK^T WMMA bursts ----
    v16bf kf[2][4];
    #pragma unroll
    for (int t = 0; t < 2; ++t) {
      const __bf16* kcol = &lk[(t * 16 + n) * KSTR + half * 16]; // contiguous k run at half*16, same as qf
      #pragma unroll
      for (int c = 0; c < 4; ++c) kf[t][c] = *(const v16bf*)(kcol + 32 * c);
    }
    v8f s[2];
    #pragma unroll
    for (int t = 0; t < 2; ++t) {
      v8f st = {};
      #pragma unroll
      for (int c = 0; c < 4; ++c)
        st = __builtin_amdgcn_wmma_f32_16x16x32_bf16(false, qf[c], false, kf[t][c],
                                                     (short)0, st, false, false);
      s[t] = st;
    }

    // ---- hoist V B-fragments: DS loads drain while softmax VALU runs ----
    v16bf vf[8];
    #pragma unroll
    for (int dc = 0; dc < 8; ++dc)
      vf[dc] = *(const v16bf*)&lvT[(dc * 16 + n) * VSTR + half * 16];

    // ---- mask add as packed v8f ops (v_pk_add_f32 / dual-issue) ----
    v8f mv0, mv1;
    {
      const float* mrowp = Msk + (size_t)(qwave + half * 8) * kS + k0 + n;
      #pragma unroll
      for (int r = 0; r < 8; ++r) {
        mv0[r] = mrowp[(size_t)r * kS];
        mv1[r] = mrowp[(size_t)r * kS + 16];
      }
    }
    v8f s0v = s[0] + mv0;
    v8f s1v = s[1] + mv1;

    // ---- online softmax (per-row reduction over 16-lane groups) ----
    v8f cv;
    #pragma unroll
    for (int r = 0; r < 8; ++r) {
      const int row = r + half * 8;                  // C layout row for VGPR r
      const float s0 = s0v[r];
      const float s1 = s1v[r];
      const float tmax = rowmax16(fmaxf(s0, s1));
      const float mnew = fmaxf(mrow[r], tmax);
      const float corr = __builtin_amdgcn_exp2f((mrow[r] - mnew) * LOG2E);
      mrow[r] = mnew;
      cv[r]   = corr;
      const float p0 = __builtin_amdgcn_exp2f((s0 - mnew) * LOG2E);
      const float p1 = __builtin_amdgcn_exp2f((s1 - mnew) * LOG2E);
      lp[w][row * PSTR + n]      = (__bf16)p0;       // C->A transpose via LDS bounce
      lp[w][row * PSTR + 16 + n] = (__bf16)p1;
      lrow[r] = lrow[r] * corr + rowsum16(p0 + p1);
    }
    // vectorized accumulator rescale: v_pk_mul_f32 pairs instead of 64 scalar muls
    #pragma unroll
    for (int dc = 0; dc < 8; ++dc) acc[dc] *= cv;

    // LDS ops are in-order per wave: safe wave-local RAW on lp
    v8bf plo = *(const v8bf*)&lp[w][n * PSTR + half * 16];
    v8bf phi = *(const v8bf*)&lp[w][n * PSTR + half * 16 + 8];
    v16bf pf = __builtin_shufflevector(plo, phi,
                 0,1,2,3,4,5,6,7,8,9,10,11,12,13,14,15);

    // ---- PV: accumulate 16x128 output, K(=keys)=32 per tile ----
    #pragma unroll
    for (int dc = 0; dc < 8; ++dc)
      acc[dc] = __builtin_amdgcn_wmma_f32_16x16x32_bf16(false, pf, false, vf[dc],
                                                        (short)0, acc[dc], false, false);
    __syncthreads();
  }

  // ---- epilogue: vector normalize, then store fp32 ----
  v8f rlv;
  #pragma unroll
  for (int r = 0; r < 8; ++r) rlv[r] = 1.0f / lrow[r];
  #pragma unroll
  for (int dc = 0; dc < 8; ++dc) acc[dc] *= rlv;
  float* orow = O + hb + (size_t)qwave * kD;
  #pragma unroll
  for (int dc = 0; dc < 8; ++dc)
    #pragma unroll
    for (int r = 0; r < 8; ++r)
      orow[(size_t)(r + half * 8) * kD + dc * 16 + n] = acc[dc][r];
}

extern "C" void kernel_launch(void* const* d_in, const int* in_sizes, int n_in,
                              void* d_out, int out_size, void* d_ws, size_t ws_size,
                              hipStream_t stream) {
  const float* q = (const float*)d_in[0];
  const float* k = (const float*)d_in[1];
  const float* v = (const float*)d_in[2];
  const float* m = (const float*)d_in[3];
  float* o = (float*)d_out;
  dim3 grid(kB * kH * (kS / QBLK));   // 512 blocks
  dim3 block(WAVES * 32);             // 8 wave32
  hipLaunchKernelGGL(fattn_bf16_wmma, grid, block, 0, stream, q, k, v, m, o);
}
